// MultiHeadSelfAttention_61065845014598
// MI455X (gfx1250) — matrix operand order repaired and checked by reading, not compiled, not run on hardware
//
#include <hip/hip_runtime.h>

// ---------------- problem constants (from reference) ----------------
#define EMB   2048
#define NH    16
#define HD    128          // head dim
#define BATCH 4
#define SEQ   2048
#define MROWS (BATCH*SEQ)  // 8192 rows of x

typedef __attribute__((ext_vector_type(16))) __bf16 v16bf;
typedef __attribute__((ext_vector_type(8)))  float  v8f;
typedef int gv4i __attribute__((vector_size(16)));  // 16B chunk for async copies

#if defined(__HIP_DEVICE_COMPILE__) && \
    __has_builtin(__builtin_amdgcn_global_load_async_to_lds_b128) && \
    __has_builtin(__builtin_amdgcn_s_wait_asynccnt)
#define HAS_ASYNC_LDS 1
#else
#define HAS_ASYNC_LDS 0
#endif

#define AS1 __attribute__((address_space(1)))
#define AS3 __attribute__((address_space(3)))

union FragU { uint4 u[2]; v16bf v; };

__device__ __forceinline__ v16bf load_frag2(const unsigned short* p0,
                                            const unsigned short* p1) {
  FragU f;
  f.u[0] = *(const uint4*)p0;
  f.u[1] = *(const uint4*)p1;
  return f.v;
}

__device__ __forceinline__ v8f vzero8() {
  v8f z;
#pragma unroll
  for (int i = 0; i < 8; ++i) z[i] = 0.0f;
  return z;
}

// round-to-nearest-even f32 -> bf16 (as raw u16)
__device__ __forceinline__ unsigned short f2bf(float f) {
  unsigned int u = __float_as_uint(f);
  u += 0x7FFFu + ((u >> 16) & 1u);
  return (unsigned short)(u >> 16);
}

__device__ __forceinline__ v8f wmma_bf16(v16bf a, v16bf b, v8f c) {
  return __builtin_amdgcn_wmma_f32_16x16x32_bf16(
      /*neg_a=*/false, a, /*neg_b=*/false, b,
      /*c_mod=*/(short)0, c, /*reuse_a=*/false, /*reuse_b=*/false);
}

template <int N>
__device__ __forceinline__ void wait_async_le() {
#if HAS_ASYNC_LDS
  __builtin_amdgcn_s_wait_asynccnt(N);  // immediate operand required
#endif
}

// ---------------- f32 -> bf16 cast ----------------
__global__ void cast_bf16_kernel(const float* __restrict__ in,
                                 unsigned short* __restrict__ out, int n) {
  int i = (blockIdx.x * blockDim.x + threadIdx.x) * 2;
  if (i < n) {
    unsigned int p = (unsigned int)f2bf(in[i]) |
                     ((unsigned int)f2bf(in[i + 1]) << 16);
    *(unsigned int*)(out + i) = p;
  }
}

// ---------------- WMMA GEMM:  C[m,n] = sum_k A[m,k] * W[n,k] ----------------
// A: [MROWS, EMB] bf16 row-major, W: [EMB, EMB] bf16 row-major (A @ W^T)
// A-tile (128 x 32 bf16) is double-buffered through LDS with async copies;
// W fragments come straight from global (weights are L2-resident on 192MB L2).
// MODE 0: store f32 C[m,n]
// MODE 1: fused RoPE epilogue, store bf16 at [b, head, s, d]   (Q / K)
// MODE 2: store bf16 transposed at [b, head, d, s]             (V)
#define AT_PAD 40  // row pitch in bf16 elems (keeps 16B align, spreads banks)

__device__ __forceinline__ void stage_A_tile(const unsigned short* __restrict__ Ablk,
                                             int k0, unsigned short (*buf)[AT_PAD],
                                             int t) {
  // 128 rows x 32 cols bf16 = 512 chunks of 16B; 256 threads x 2 chunks
#pragma unroll
  for (int j = 0; j < 2; ++j) {
    int c = t + j * 256;
    int row = c >> 2;
    int grp = c & 3;
    const unsigned short* gp = Ablk + (size_t)row * EMB + k0 + grp * 8;
    unsigned short* lp = &buf[row][grp * 8];
#if HAS_ASYNC_LDS
    __builtin_amdgcn_global_load_async_to_lds_b128(
        (AS1 gv4i*)gp, (AS3 gv4i*)lp, 0, 0);
#else
    *(uint4*)lp = *(const uint4*)gp;
#endif
  }
}

template <int MODE>
__global__ __launch_bounds__(256) void gemm_wmma(
    const unsigned short* __restrict__ A, const unsigned short* __restrict__ W,
    float* __restrict__ Cf, unsigned short* __restrict__ Cbf) {
  __shared__ __align__(16) unsigned short Atile[2][128][AT_PAD];

  const int K = EMB;
  const int lane = threadIdx.x & 31;
  const int wv   = threadIdx.x >> 5;   // 8 waves
  const int wm   = wv & 3;             // 4 waves along M
  const int wn   = wv >> 2;            // 2 waves along N
  const int m0 = blockIdx.y * 128 + wm * 32;   // wave: 32 rows
  const int n0 = blockIdx.x * 128 + wn * 64;   // wave: 64 cols
  const int r = lane & 15;
  const int h = lane >> 4;

  v8f acc[2][4];
#pragma unroll
  for (int mi = 0; mi < 2; ++mi)
#pragma unroll
    for (int ni = 0; ni < 4; ++ni) acc[mi][ni] = vzero8();

  size_t boff[4];
#pragma unroll
  for (int ni = 0; ni < 4; ++ni)
    boff[ni] = (size_t)(n0 + ni * 16 + r) * K + h * 8;  // B lane layout

  const unsigned short* Ablk = A + (size_t)(blockIdx.y * 128) * K;
  const int t = threadIdx.x;

  // prologue: stage first A slab
  stage_A_tile(Ablk, 0, Atile[0], t);
  int cur = 0;

  for (int k0 = 0; k0 < K; k0 += 32) {
    const bool more = (k0 + 32) < K;
    if (more) {
      stage_A_tile(Ablk, k0 + 32, Atile[cur ^ 1], t);  // overlap next slab
      wait_async_le<2>();  // my 2 copies of the *current* slab have landed
    } else {
      wait_async_le<0>();
    }
    __syncthreads();  // everyone's copies visible

    if (more) {  // keep weight stream warm in L2
      __builtin_prefetch(W + boff[0] + k0 + 32, 0, 1);
      __builtin_prefetch(W + boff[2] + k0 + 32, 0, 1);
    }

    v16bf af[2], bf[4];
#pragma unroll
    for (int mi = 0; mi < 2; ++mi) {
      const unsigned short* ap = &Atile[cur][wm * 32 + mi * 16 + r][h * 8];
      af[mi] = load_frag2(ap, ap + 16);  // ds_load_b128 x2, A lane layout
    }
#pragma unroll
    for (int ni = 0; ni < 4; ++ni)
      bf[ni] = load_frag2(W + boff[ni] + k0, W + boff[ni] + k0 + 16);
#pragma unroll
    for (int mi = 0; mi < 2; ++mi)
#pragma unroll
      for (int ni = 0; ni < 4; ++ni)
        acc[mi][ni] = wmma_bf16(af[mi], bf[ni], acc[mi][ni]);

    __syncthreads();  // WAR: slab may be overwritten next iteration
    cur ^= 1;
  }

  // epilogue; C layout: VGPR v -> row (v + 8*h), lane r -> col
#pragma unroll
  for (int mi = 0; mi < 2; ++mi) {
#pragma unroll
    for (int ni = 0; ni < 4; ++ni) {
#pragma unroll
      for (int v = 0; v < 8; ++v) {
        int row = m0 + mi * 16 + v + 8 * h;   // global m = b*SEQ + s
        int col = n0 + ni * 16 + r;           // emb column
        float val = acc[mi][ni][v];
        if (MODE == 0) {
          Cf[(size_t)row * EMB + col] = val;
        } else {
          int b = row >> 11;           // / SEQ
          int spos = row & (SEQ - 1);
          int head = col >> 7;         // / HD
          int d = col & (HD - 1);
          if (MODE == 1) {
            // RoPE: pair (2i, 2i+1) lives in adjacent lanes of this fragment
            float other = __shfl_xor(val, 1, 32);
            float i2 = (float)(d & ~1);
            float ang = (float)spos * __expf(i2 * (-9.210340371976184f / 128.0f));
            float sn, cs;
            __sincosf(ang, &sn, &cs);
            float outv = (d & 1) ? (val * cs + other * sn)
                                 : (val * cs - other * sn);
            Cbf[((size_t)(b * NH + head) * SEQ + spos) * HD + d] = f2bf(outv);
          } else {  // MODE == 2 : V, store transposed [b,h,d,s]
            Cbf[((size_t)(b * NH + head) * HD + d) * SEQ + spos] = f2bf(val);
          }
        }
      }
    }
  }
}

// ---------------- flash attention (causal, online softmax) ----------------
// Qb/Kb: [B*NH, SEQ, HD] bf16,  Vt: [B*NH, HD, SEQ] bf16, Ctx: [B*SEQ, EMB] bf16
__global__ __launch_bounds__(128) void attn_kernel(
    const unsigned short* __restrict__ Qb, const unsigned short* __restrict__ Kb,
    const unsigned short* __restrict__ Vt, unsigned short* __restrict__ Ctx) {
  __shared__ __align__(16) unsigned short ldsP[4][16][32];  // per-wave P tile

  const int lane = threadIdx.x & 31;
  const int wv   = threadIdx.x >> 5;  // 4 waves, 16 query rows each
  const int r = lane & 15;
  const int h = lane >> 4;

  const int qchunks = SEQ / 64;
  const int bh   = blockIdx.x / qchunks;
  const int qblk = blockIdx.x % qchunks;
  const int q0 = qblk * 64 + wv * 16;

  const unsigned short* Qbh = Qb + (size_t)bh * SEQ * HD;
  const unsigned short* Kbh = Kb + (size_t)bh * SEQ * HD;
  const unsigned short* Vbh = Vt + (size_t)bh * HD * SEQ;

  // Q fragments (16 rows x 128, as 4 A-frags of K=32)
  v16bf qf[4];
#pragma unroll
  for (int c = 0; c < 4; ++c) {
    const unsigned short* p = Qbh + (size_t)(q0 + r) * HD + c * 32 + h * 8;
    qf[c] = load_frag2(p, p + 16);
  }

  float mrow[8], lrow[8];
  v8f o[8];
#pragma unroll
  for (int v = 0; v < 8; ++v) { mrow[v] = -__builtin_inff(); lrow[v] = 0.0f; }
#pragma unroll
  for (int dc = 0; dc < 8; ++dc) o[dc] = vzero8();

  const int roff = 8 * h;                 // C-frag row = q0 + v + roff
  const int kbmax = (q0 + 15) >> 5;       // causal: last 32-key block needed
  const float scale = 0.08838834764831845f;  // 1/sqrt(128)

  for (int kb = 0; kb <= kbmax; ++kb) {
    const int kbase = kb * 32;
    float sv[2][8];
#pragma unroll
    for (int t = 0; t < 2; ++t) {
      v8f accs = vzero8();
      const unsigned short* kp =
          Kbh + (size_t)(kbase + t * 16 + r) * HD + h * 8;  // B-frag: key=r
#pragma unroll
      for (int c = 0; c < 4; ++c) {
        v16bf kf = load_frag2(kp + c * 32, kp + c * 32 + 16);
        accs = wmma_bf16(qf[c], kf, accs);
      }
#pragma unroll
      for (int v = 0; v < 8; ++v) {
        int row = q0 + v + roff;
        int col = kbase + t * 16 + r;
        sv[t][v] = (col <= row) ? accs[v] * scale : -__builtin_inff();
      }
    }
    // online softmax update (rows live across 16 lanes of each half)
#pragma unroll
    for (int v = 0; v < 8; ++v) {
      float rm = fmaxf(sv[0][v], sv[1][v]);
      rm = fmaxf(rm, __shfl_xor(rm, 1, 32));
      rm = fmaxf(rm, __shfl_xor(rm, 2, 32));
      rm = fmaxf(rm, __shfl_xor(rm, 4, 32));
      rm = fmaxf(rm, __shfl_xor(rm, 8, 32));
      float mnew = fmaxf(mrow[v], rm);
      float alpha, p0, p1;
      if (mnew == -__builtin_inff()) {  // fully-masked block for this row
        alpha = 1.0f; p0 = 0.0f; p1 = 0.0f;
      } else {
        alpha = __expf(mrow[v] - mnew);
        p0 = __expf(sv[0][v] - mnew);
        p1 = __expf(sv[1][v] - mnew);
      }
      mrow[v] = mnew;
      float rs = p0 + p1;
      rs += __shfl_xor(rs, 1, 32);
      rs += __shfl_xor(rs, 2, 32);
      rs += __shfl_xor(rs, 4, 32);
      rs += __shfl_xor(rs, 8, 32);
      lrow[v] = lrow[v] * alpha + rs;
#pragma unroll
      for (int dc = 0; dc < 8; ++dc) o[dc][v] *= alpha;
      int prow = v + roff;
      ldsP[wv][prow][r]      = f2bf(p0);
      ldsP[wv][prow][16 + r] = f2bf(p1);
    }
    // LDS store -> load reshuffle (C layout -> A layout), same-wave ordering
    asm volatile("s_wait_dscnt 0" ::: "memory");
    const unsigned short* pp = &ldsP[wv][r][h * 8];
    v16bf pf = load_frag2(pp, pp + 16);
    // O += P (16x32) x V (32x16 per d-chunk); Vt rows are contiguous in s
#pragma unroll
    for (int dc = 0; dc < 8; ++dc) {
      const unsigned short* vp =
          Vbh + (size_t)(dc * 16 + r) * SEQ + kbase + h * 8;
      v16bf vf = load_frag2(vp, vp + 16);
      o[dc] = wmma_bf16(pf, vf, o[dc]);
    }
  }

  // normalize + store ctx as bf16 [b*SEQ+s, EMB]
  const int b = bh / NH, head = bh % NH;
#pragma unroll
  for (int v = 0; v < 8; ++v) {
    float inv = 1.0f / lrow[v];
    int spos = q0 + v + roff;
#pragma unroll
    for (int dc = 0; dc < 8; ++dc) {
      int col = head * HD + dc * 16 + r;
      Ctx[(size_t)(b * SEQ + spos) * EMB + col] = f2bf(o[dc][v] * inv);
    }
  }
}

// ---------------- launcher ----------------
extern "C" void kernel_launch(void* const* d_in, const int* in_sizes, int n_in,
                              void* d_out, int out_size, void* d_ws,
                              size_t ws_size, hipStream_t stream) {
  const float* x  = (const float*)d_in[0];
  const float* Wq = (const float*)d_in[1];
  const float* Wk = (const float*)d_in[2];
  const float* Wv = (const float*)d_in[3];
  const float* Wo = (const float*)d_in[4];
  float* out = (float*)d_out;

  const size_t NX = (size_t)MROWS * EMB;  // 16.8M elems
  const size_t NW = (size_t)EMB * EMB;    //  4.2M elems

  unsigned short* xbf  = (unsigned short*)d_ws;
  unsigned short* wqbf = xbf + NX;
  unsigned short* wkbf = wqbf + NW;
  unsigned short* wvbf = wkbf + NW;
  unsigned short* wobf = wvbf + NW;
  unsigned short* qbf  = wobf + NW;   // [B,H,S,D]
  unsigned short* kbf  = qbf + NX;    // [B,H,S,D]
  unsigned short* vtbf = kbf + NX;    // [B,H,D,S]
  unsigned short* ctx  = vtbf + NX;   // [B*S, EMB]
  // total ws usage: (NX*5 + NW*4) * 2 bytes ~= 201 MB

  // 1) bf16 casts
  cast_bf16_kernel<<<(int)(NX / 512), 256, 0, stream>>>(x, xbf, (int)NX);
  cast_bf16_kernel<<<(int)(NW / 512), 256, 0, stream>>>(Wq, wqbf, (int)NW);
  cast_bf16_kernel<<<(int)(NW / 512), 256, 0, stream>>>(Wk, wkbf, (int)NW);
  cast_bf16_kernel<<<(int)(NW / 512), 256, 0, stream>>>(Wv, wvbf, (int)NW);
  cast_bf16_kernel<<<(int)(NW / 512), 256, 0, stream>>>(Wo, wobf, (int)NW);

  // 2) QKV projections with fused RoPE / transpose epilogues
  dim3 ggrid(EMB / 128, MROWS / 128);
  gemm_wmma<1><<<ggrid, 256, 0, stream>>>(xbf, wqbf, nullptr, qbf);
  gemm_wmma<1><<<ggrid, 256, 0, stream>>>(xbf, wkbf, nullptr, kbf);
  gemm_wmma<2><<<ggrid, 256, 0, stream>>>(xbf, wvbf, nullptr, vtbf);

  // 3) causal flash attention
  attn_kernel<<<BATCH * NH * (SEQ / 64), 128, 0, stream>>>(qbf, kbf, vtbf, ctx);

  // 4) output projection (f32 result)
  gemm_wmma<0><<<ggrid, 256, 0, stream>>>(ctx, wobf, out, nullptr);
}
